// GlobalM_GRS_292057776838
// MI455X (gfx1250) — matrix operand order repaired and checked by reading, not compiled, not run on hardware
//
#include <hip/hip_runtime.h>

// ---------------------------------------------------------------------------
// Types for CDNA5 WMMA bf16 path + TDM descriptors
// ---------------------------------------------------------------------------
typedef __attribute__((ext_vector_type(16))) __bf16         v16bf;
typedef __attribute__((ext_vector_type(16))) unsigned short v16us;
typedef __attribute__((ext_vector_type(8)))  float          v8f;
typedef __attribute__((ext_vector_type(4)))  unsigned int   v4u;
typedef __attribute__((ext_vector_type(8)))  int            v8i;
typedef __attribute__((ext_vector_type(4)))  int            v4i;

#if defined(__has_builtin)
#  if __has_builtin(__builtin_amdgcn_tensor_load_to_lds)
#    define HAVE_TDM 1
#  endif
#endif
#ifndef HAVE_TDM
#  define HAVE_TDM 0
#endif

static __device__ __forceinline__ unsigned short f2us(float f) {
  // f32 -> bf16 round-to-nearest-even
  unsigned int u = __builtin_bit_cast(unsigned int, f);
  u += 0x7FFFu + ((u >> 16) & 1u);
  return (unsigned short)(u >> 16);
}

static __device__ __forceinline__ v8f vzero() {
  v8f z;
#pragma unroll
  for (int i = 0; i < 8; ++i) z[i] = 0.0f;
  return z;
}

static __device__ __forceinline__ v8f wmma_bf16(v16bf a, v16bf b, v8f c) {
  return __builtin_amdgcn_wmma_f32_16x16x32_bf16(
      /*neg_a=*/false, a, /*neg_b=*/false, b,
      /*c_mod=*/(short)0, c, /*reuse_a=*/false, /*reuse_b=*/false);
}

#if HAVE_TDM
// 1-D contiguous TDM copy: global (bf16 elements) -> LDS.
// D# per CDNA5 ISA ch.8: group0 {count=1, lds_addr, global_addr, type=2},
// group1 {data_size=2B, tensor_dim0=tile_dim0=nelem, tensor_dim1=tile_dim1=1,
//         dim0_stride=nelem}.  Groups 2/3 zero (<=2D tensor).
static __device__ __forceinline__ void
tdm_load_1d(void* lds_dst, const void* gsrc, unsigned nelem) {
  const unsigned lds_off = (unsigned)(size_t)lds_dst;            // LDS byte offset
  const unsigned long long ga = (unsigned long long)(size_t)gsrc;
  v4u g0;
  g0[0] = 1u;                                                    // count=1 (user D#)
  g0[1] = lds_off;                                               // lds_addr
  g0[2] = (unsigned)(ga & 0xFFFFFFFFu);                          // global_addr lo
  g0[3] = (unsigned)((ga >> 32) & 0x01FFFFFFu) | 0x80000000u;    // addr hi | type=2
  v8i g1;
  g1[0] = 0x10000;                                 // data_size=1 -> 2 bytes
  g1[1] = (int)((nelem & 0xFFFFu) << 16);          // tensor_dim0[15:0] @ bits63:48
  g1[2] = (int)(((nelem >> 16) & 0xFFFFu) | (1u << 16)); // dim0 hi | tensor_dim1=1
  g1[3] = (int)((nelem & 0xFFFFu) << 16);          // tile_dim0 @ bits127:112
  g1[4] = 1;                                       // tile_dim1=1, tile_dim2=0
  g1[5] = (int)nelem;                              // tensor_dim0_stride lo32
  g1[6] = 0;
  g1[7] = 0;
  v4i gz = {0, 0, 0, 0};
#if __clang_major__ >= 23
  v8i gz8 = {0, 0, 0, 0, 0, 0, 0, 0};
  __builtin_amdgcn_tensor_load_to_lds(g0, g1, gz, gz, gz8, 0);
#else
  __builtin_amdgcn_tensor_load_to_lds(g0, g1, gz, gz, 0);
#endif
}
#endif  // HAVE_TDM

// A-matrix fragment (16 x 32, bf16) from M-major LDS tile (ld = K stride).
static __device__ __forceinline__ v16bf
load_a(const unsigned short* p, int ld, int mbase, int kbase) {
  const int lane = threadIdx.x & 31;
  const int h    = lane >> 4;
  const unsigned short* row = p + (size_t)(mbase + (lane & 15)) * ld + kbase;
  v16us a;
#pragma unroll
  for (int v = 0; v < 4; ++v) {
    const int k = h * 8 + 2 * v;
    a[2 * v]         = row[k];
    a[2 * v + 1]     = row[k + 1];
    a[8 + 2 * v]     = row[16 + k];
    a[8 + 2 * v + 1] = row[16 + k + 1];
  }
  return __builtin_bit_cast(v16bf, a);
}

// B-matrix fragment (32 x 16, bf16) from an N-MAJOR LDS tile T (T[n][k]=B[k][n]).
static __device__ __forceinline__ v16bf
load_b_nm(const unsigned short* p, int ld, int kbase, int nbase) {
  const int lane = threadIdx.x & 31;
  const int h    = lane >> 4;
  const unsigned short* row = p + (size_t)(nbase + (lane & 15)) * ld + kbase;
  v16us b;
#pragma unroll
  for (int v = 0; v < 8; ++v) {
    const int k = h * 8 + (v & 3) * 2 + (v >> 2) * 16;
    b[2 * v]     = row[k];
    b[2 * v + 1] = row[k + 1];
  }
  return __builtin_bit_cast(v16bf, b);
}

// ---------------------------------------------------------------------------
// Problem constants
// ---------------------------------------------------------------------------
#define BB   8
#define CC   256
#define HH   128
#define WW   128
#define PIX  (HH * WW)        // 16384
#define HID  256
#define NTHR 256

#define VT_LD  136            // V^T tile leading dim (pad: 128+8)
#define XT_LD  264            // MLP activation^T leading dim (pad: 256+8)

// ---------------------------------------------------------------------------
// Weight pre-conversion: f32 -> bf16 in per-chunk LDS order (once per call)
// ---------------------------------------------------------------------------
__global__ void cvt_conv_w_kernel(const float* __restrict__ w,
                                  unsigned short* __restrict__ out) {
  const int i = blockIdx.x * NTHR + threadIdx.x;    // 0 .. 589823
  if (i >= 9 * CC * CC) return;
  const int blk = i >> 13;                           // (kh*8+cc)*3+kw
  const int r   = i & 8191;
  const int co  = r >> 5;
  const int ci  = r & 31;
  const int kw  = blk % 3;
  const int t   = blk / 3;
  const int cc2 = t & 7;
  const int kh  = t >> 3;
  out[i] = f2us(w[(((size_t)co * CC + cc2 * 32 + ci) * 3 + kh) * 3 + kw]);
}

__global__ void cvt_mlp_w_kernel(const float* __restrict__ w1,
                                 const float* __restrict__ w2,
                                 unsigned short* __restrict__ o1,
                                 unsigned short* __restrict__ o2) {
  const int i = blockIdx.x * NTHR + threadIdx.x;    // 0 .. 65535
  if (i >= CC * HID) return;
  const int kc = i >> 13;
  const int r  = i & 8191;
  const int m  = r >> 5;
  const int kl = r & 31;
  o1[i] = f2us(w1[(size_t)m * CC + kc * 32 + kl]);
  o2[i] = f2us(w2[(size_t)m * HID + kc * 32 + kl]);
}

// ---------------------------------------------------------------------------
// LayerNorm over the pixel dim (16384) per (b,c)
// ---------------------------------------------------------------------------
template <bool OUT_BF16>
__global__ void ln_kernel(const float* __restrict__ in, void* __restrict__ outv) {
  __shared__ float red[NTHR];
  const int tid = threadIdx.x;
  const float* p = in + (size_t)blockIdx.x * PIX;

  float s = 0.f;
  for (int i = tid; i < PIX; i += NTHR) s += p[i];
  red[tid] = s;
  __syncthreads();
  for (int o = NTHR / 2; o > 0; o >>= 1) {
    if (tid < o) red[tid] += red[tid + o];
    __syncthreads();
  }
  const float mu = red[0] * (1.0f / PIX);
  __syncthreads();

  float s2 = 0.f;
  for (int i = tid; i < PIX; i += NTHR) { float d = p[i] - mu; s2 += d * d; }
  red[tid] = s2;
  __syncthreads();
  for (int o = NTHR / 2; o > 0; o >>= 1) {
    if (tid < o) red[tid] += red[tid + o];
    __syncthreads();
  }
  const float inv = rsqrtf(red[0] * (1.0f / PIX) + 1e-5f);

  if (OUT_BF16) {
    unsigned short* q = (unsigned short*)outv + (size_t)blockIdx.x * PIX;
    for (int i = tid; i < PIX; i += NTHR) q[i] = f2us((p[i] - mu) * inv);
  } else {
    float* q = (float*)outv + (size_t)blockIdx.x * PIX;
    for (int i = tid; i < PIX; i += NTHR) q[i] = (p[i] - mu) * inv;
  }
}

// ---------------------------------------------------------------------------
// Per-(b,c) row attention (unchanged from round 2 -- already b128-clean)
// ---------------------------------------------------------------------------
__global__ void attn_kernel(const float* __restrict__ x0,
                            const float* __restrict__ wq, const float* __restrict__ bq,
                            const float* __restrict__ wk, const float* __restrict__ bk,
                            const float* __restrict__ wv, const float* __restrict__ bv,
                            unsigned short* __restrict__ outp) {
  extern __shared__ unsigned char smem[];
  unsigned short* Qs = (unsigned short*)smem;           // [128][128]
  unsigned short* Ks = Qs + PIX;                        // [128][128]
  unsigned short* Vt = Ks + PIX;                        // [128 w][VT_LD g]
  float*          Ss = (float*)(Vt + HH * VT_LD);       // [8][16][128]
  unsigned short* Ps = (unsigned short*)(Ss + 8 * 16 * WW); // [8][16][128]

  const int bc = blockIdx.x;
  const int c  = bc & (CC - 1);
  const float scale = rsqrtf((float)WW);
  const float sq = wq[c] * scale, oq = bq[c] * scale;
  const float sk = wk[c],         ok = bk[c];
  const float sv = wv[c],         ov = bv[c];

  const float* xp = x0 + (size_t)bc * PIX;
  for (int i = threadIdx.x; i < PIX; i += NTHR) {
    const float v = xp[i];
    const int g = i >> 7, w = i & (WW - 1);
    Qs[i] = f2us(v * sq + oq);
    Ks[i] = f2us(v * sk + ok);
    Vt[(size_t)w * VT_LD + g] = f2us(v * sv + ov);     // transpose V
  }
  __syncthreads();

  const int wave  = threadIdx.x >> 5;
  const int lane  = threadIdx.x & 31;
  const int mbase = wave * 16;
  float*          Sw = Ss + (size_t)wave * 16 * WW;
  unsigned short* Pw = Ps + (size_t)wave * 16 * WW;

  // S = Q * K^T
  for (int nt = 0; nt < 8; ++nt) {
    v8f acc = vzero();
#pragma unroll
    for (int kc = 0; kc < 4; ++kc) {
      v16bf a = load_a   (Qs, WW, mbase, kc * 32);
      v16bf b = load_b_nm(Ks, WW, kc * 32, nt * 16);
      acc = wmma_bf16(a, b, acc);
    }
    const int h = lane >> 4, n = lane & 15;
#pragma unroll
    for (int r = 0; r < 8; ++r)
      Sw[(h * 8 + r) * WW + nt * 16 + n] = acc[r];
  }
  __syncthreads();

  // row softmax -> P (bf16)
  if (lane < 16) {
    float* row = Sw + lane * WW;
    float mx = -3.0e38f;
    for (int j = 0; j < WW; ++j) mx = fmaxf(mx, row[j]);
    float sum = 0.f;
    for (int j = 0; j < WW; ++j) { const float e = __expf(row[j] - mx); row[j] = e; sum += e; }
    const float invs = 1.0f / sum;
    unsigned short* pr = Pw + lane * WW;
    for (int j = 0; j < WW; ++j) pr[j] = f2us(row[j] * invs);
  }
  __syncthreads();

  // O = P * V
  unsigned short* op = outp + (size_t)bc * PIX;
  for (int nt = 0; nt < 8; ++nt) {
    v8f acc = vzero();
#pragma unroll
    for (int kc = 0; kc < 4; ++kc) {
      v16bf a = load_a   (Pw, WW, 0, kc * 32);
      v16bf b = load_b_nm(Vt, VT_LD, kc * 32, nt * 16);
      acc = wmma_bf16(a, b, acc);
    }
    const int h = lane >> 4, n = lane & 15;
#pragma unroll
    for (int r = 0; r < 8; ++r)
      op[(size_t)(mbase + h * 8 + r) * WW + nt * 16 + n] = f2us(acc[r]);
  }
}

// ---------------------------------------------------------------------------
// 3x3 SAME conv (256->256) + residual, implicit GEMM.
// Weight chunks (48KB flat blocks) streamed into LDS by the Tensor Data
// Mover, DOUBLE-BUFFERED: issue TDM for chunk j+1, stage input rows for
// chunk j (manual: needs transpose+pad), s_wait_tensorcnt, barrier, compute.
// ---------------------------------------------------------------------------
__global__ void conv3x3_res_kernel(const unsigned short* __restrict__ src,
                                   const float* __restrict__ xres,
                                   const unsigned short* __restrict__ wcvt,
                                   float* __restrict__ xnew) {
  extern __shared__ unsigned char smem[];
  unsigned short* Aw0 = (unsigned short*)smem;  // [3 kw][256 co][32 ci]
  unsigned short* Aw1 = Aw0 + 3 * 256 * 32;     // double buffer
  unsigned short* Xt  = Aw1 + 3 * 256 * 32;     // [130 w_pad][32 ci]

  const int bh = blockIdx.x;
  const int b  = bh >> 7;
  const int h  = bh & (HH - 1);
  const int wave = threadIdx.x >> 5;
  const int lane = threadIdx.x & 31;

  // valid kh range (uniform per block) -> linear chunk sequence
  const int kh0  = (h == 0) ? 1 : 0;
  const int kh1  = (h == HH - 1) ? 1 : 2;
  const int nblk = (kh1 - kh0 + 1) * 8;
  unsigned short* Abuf[2] = {Aw0, Aw1};

  v8f acc[2][8];
#pragma unroll
  for (int mt = 0; mt < 2; ++mt)
#pragma unroll
    for (int nt = 0; nt < 8; ++nt) acc[mt][nt] = vzero();

#if HAVE_TDM
  if (threadIdx.x < 32)   // one wave drives the DMA
    tdm_load_1d(Abuf[0], wcvt + (size_t)(kh0 * 8) * 24576, 24576);
#endif

  for (int j = 0; j < nblk; ++j) {
    const int kh     = kh0 + (j >> 3);
    const int cc     = j & 7;
    const int row    = h + kh - 1;
    const int cibase = cc * 32;
    __syncthreads();                       // Abuf[(j+1)&1] and Xt free

#if HAVE_TDM
    if (threadIdx.x < 32) {
      if (j + 1 < nblk) {
        const int khn = kh0 + ((j + 1) >> 3);
        const int ccn = (j + 1) & 7;
        tdm_load_1d(Abuf[(j + 1) & 1],
                    wcvt + (size_t)(khn * 8 + ccn) * 24576, 24576);
        // prefetch next chunk's input rows (32 rows, one per lane)
        const int rown = h + khn - 1;
        __builtin_prefetch(
            src + (((size_t)b * CC + ccn * 32 + lane) * HH + rown) * WW, 0, 0);
      }
    }
#else
    {
      const uint4* gsrc = (const uint4*)(wcvt + (size_t)(kh * 8 + cc) * 24576);
      uint4* adst = (uint4*)Abuf[j & 1];
      for (int i = threadIdx.x; i < 3072; i += NTHR) adst[i] = gsrc[i];
    }
#endif

    // stage zero-padded input row chunk, transposed to [w_pad][ci]
    for (int i = threadIdx.x; i < 32 * 130; i += NTHR) {
      const int ci = i / 130;
      const int wp = i - ci * 130;
      const int wc = wp - 1;
      unsigned short v = 0;
      if (wc >= 0 && wc < WW)
        v = src[(((size_t)b * CC + cibase + ci) * HH + row) * WW + wc];
      Xt[wp * 32 + ci] = v;
    }

#if HAVE_TDM
    if (threadIdx.x < 32) {                 // chunk j's DMA must be complete
      if (j + 1 < nblk) __builtin_amdgcn_s_wait_tensorcnt(1);
      else              __builtin_amdgcn_s_wait_tensorcnt(0);
    }
#endif
    __syncthreads();

    const unsigned short* Awc = Abuf[j & 1];
#pragma unroll
    for (int kw = 0; kw < 3; ++kw) {
      const unsigned short* Ak = Awc + kw * (256 * 32);
      v16bf a0 = load_a(Ak, 32, (wave * 2 + 0) * 16, 0);
      v16bf a1 = load_a(Ak, 32, (wave * 2 + 1) * 16, 0);
#pragma unroll
      for (int nt = 0; nt < 8; ++nt) {
        v16bf bf = load_b_nm(Xt, 32, 0, nt * 16 + kw);
        acc[0][nt] = wmma_bf16(a0, bf, acc[0][nt]);
        acc[1][nt] = wmma_bf16(a1, bf, acc[1][nt]);
      }
    }
  }
  __syncthreads();

  const int hh = lane >> 4, n = lane & 15;
#pragma unroll
  for (int mt = 0; mt < 2; ++mt) {
    const int cobase = (wave * 2 + mt) * 16;
#pragma unroll
    for (int nt = 0; nt < 8; ++nt) {
#pragma unroll
      for (int r = 0; r < 8; ++r) {
        const int co = cobase + hh * 8 + r;
        const int wc = nt * 16 + n;
        const size_t idx = (((size_t)b * CC + co) * HH + h) * WW + wc;
        xnew[idx] = acc[mt][nt][r] + xres[idx];
      }
    }
  }
}

// ---------------------------------------------------------------------------
// conv-MLP: y = W2 * relu(W1 * xn2) + x_new.  Weight chunks via TDM.
// ---------------------------------------------------------------------------
__global__ void mlp_kernel(const unsigned short* __restrict__ xn2,
                           const float* __restrict__ xnew,
                           const unsigned short* __restrict__ w1c,
                           const unsigned short* __restrict__ w2c,
                           float* __restrict__ out) {
  extern __shared__ unsigned char smem[];
  unsigned short* Xt = (unsigned short*)smem;   // [128 w][XT_LD c]
  unsigned short* Ht = Xt + HH * XT_LD;         // [128 w][XT_LD d]
  unsigned short* Ws = Ht + HH * XT_LD;         // [256 m][32 k] chunk

  const int bh = blockIdx.x;
  const int b  = bh >> 7;
  const int h  = bh & (HH - 1);
  const int wave = threadIdx.x >> 5;
  const int lane = threadIdx.x & 31;
  const int hh = lane >> 4, n = lane & 15;

  for (int i = threadIdx.x; i < CC * WW; i += NTHR) {
    const int c = i >> 7, w = i & (WW - 1);
    Xt[(size_t)w * XT_LD + c] = xn2[(((size_t)b * CC + c) * HH + h) * WW + w];
  }

  v8f acc[2][8];
#pragma unroll
  for (int mt = 0; mt < 2; ++mt)
#pragma unroll
    for (int nt = 0; nt < 8; ++nt) acc[mt][nt] = vzero();

  // GEMM1: h = relu(W1 x)
  for (int kc = 0; kc < 8; ++kc) {
    __syncthreads();
#if HAVE_TDM
    if (threadIdx.x < 32) {
      tdm_load_1d(Ws, w1c + (size_t)kc * 8192, 8192);
      __builtin_amdgcn_s_wait_tensorcnt(0);
    }
#else
    {
      const uint4* gsrc = (const uint4*)(w1c + (size_t)kc * 8192);
      uint4* adst = (uint4*)Ws;
      for (int i = threadIdx.x; i < 1024; i += NTHR) adst[i] = gsrc[i];
    }
#endif
    __syncthreads();
    v16bf a0 = load_a(Ws, 32, (wave * 2 + 0) * 16, 0);
    v16bf a1 = load_a(Ws, 32, (wave * 2 + 1) * 16, 0);
#pragma unroll
    for (int nt = 0; nt < 8; ++nt) {
      v16bf bf = load_b_nm(Xt, XT_LD, kc * 32, nt * 16);
      acc[0][nt] = wmma_bf16(a0, bf, acc[0][nt]);
      acc[1][nt] = wmma_bf16(a1, bf, acc[1][nt]);
    }
  }
  __syncthreads();
#pragma unroll
  for (int mt = 0; mt < 2; ++mt) {
    const int dbase = (wave * 2 + mt) * 16;
#pragma unroll
    for (int nt = 0; nt < 8; ++nt)
#pragma unroll
      for (int r = 0; r < 8; ++r)
        Ht[(size_t)(nt * 16 + n) * XT_LD + dbase + hh * 8 + r] =
            f2us(fmaxf(acc[mt][nt][r], 0.0f));
  }
  __syncthreads();

  // GEMM2: y = W2 h
#pragma unroll
  for (int mt = 0; mt < 2; ++mt)
#pragma unroll
    for (int nt = 0; nt < 8; ++nt) acc[mt][nt] = vzero();

  for (int kc = 0; kc < 8; ++kc) {
    __syncthreads();
#if HAVE_TDM
    if (threadIdx.x < 32) {
      tdm_load_1d(Ws, w2c + (size_t)kc * 8192, 8192);
      __builtin_amdgcn_s_wait_tensorcnt(0);
    }
#else
    {
      const uint4* gsrc = (const uint4*)(w2c + (size_t)kc * 8192);
      uint4* adst = (uint4*)Ws;
      for (int i = threadIdx.x; i < 1024; i += NTHR) adst[i] = gsrc[i];
    }
#endif
    __syncthreads();
    v16bf a0 = load_a(Ws, 32, (wave * 2 + 0) * 16, 0);
    v16bf a1 = load_a(Ws, 32, (wave * 2 + 1) * 16, 0);
#pragma unroll
    for (int nt = 0; nt < 8; ++nt) {
      v16bf bf = load_b_nm(Ht, XT_LD, kc * 32, nt * 16);
      acc[0][nt] = wmma_bf16(a0, bf, acc[0][nt]);
      acc[1][nt] = wmma_bf16(a1, bf, acc[1][nt]);
    }
  }
  __syncthreads();

#pragma unroll
  for (int mt = 0; mt < 2; ++mt) {
    const int cbase = (wave * 2 + mt) * 16;
#pragma unroll
    for (int nt = 0; nt < 8; ++nt)
#pragma unroll
      for (int r = 0; r < 8; ++r) {
        const int c = cbase + hh * 8 + r;
        const int w = nt * 16 + n;
        const size_t idx = (((size_t)b * CC + c) * HH + h) * WW + w;
        out[idx] = acc[mt][nt][r] + xnew[idx];
      }
  }
}

// ---------------------------------------------------------------------------
// Host launcher
// ---------------------------------------------------------------------------
extern "C" void kernel_launch(void* const* d_in, const int* in_sizes, int n_in,
                              void* d_out, int out_size, void* d_ws, size_t ws_size,
                              hipStream_t stream) {
  const float* x       = (const float*)d_in[0];
  const float* wq      = (const float*)d_in[1];
  const float* bq      = (const float*)d_in[2];
  const float* wk      = (const float*)d_in[3];
  const float* bk      = (const float*)d_in[4];
  const float* wv      = (const float*)d_in[5];
  const float* bv      = (const float*)d_in[6];
  const float* w_inter = (const float*)d_in[7];
  const float* w_mlp1  = (const float*)d_in[8];
  const float* w_mlp2  = (const float*)d_in[9];
  float* out = (float*)d_out;

  const size_t N = (size_t)BB * CC * HH * WW;          // 33,554,432
  float*          buf0  = (float*)d_ws;                // x0 (LN1), then x_new
  unsigned short* buf1  = (unsigned short*)(buf0 + N); // attn bf16, then xn2 bf16
  unsigned short* wconv = buf1 + N;                    // 589,824 us
  unsigned short* w1c   = wconv + 9 * CC * CC;         // 65,536 us
  unsigned short* w2c   = w1c + CC * HID;              // 65,536 us

  const int attn_smem = (PIX + PIX + HH * VT_LD + PIX) * 2 + 8 * 16 * WW * 4; // 198656
  const int conv_smem = (2 * 3 * 256 * 32 + 130 * 32) * 2;                    // 106624
  const int mlp_smem  = (2 * HH * XT_LD + 256 * 32) * 2;                      // 151552

  hipFuncSetAttribute((const void*)attn_kernel,
                      hipFuncAttributeMaxDynamicSharedMemorySize, attn_smem);
  hipFuncSetAttribute((const void*)conv3x3_res_kernel,
                      hipFuncAttributeMaxDynamicSharedMemorySize, conv_smem);
  hipFuncSetAttribute((const void*)mlp_kernel,
                      hipFuncAttributeMaxDynamicSharedMemorySize, mlp_smem);

  // 0) one-time (per call) weight conversion to bf16 in LDS-chunk order
  cvt_conv_w_kernel<<<(9 * CC * CC + NTHR - 1) / NTHR, NTHR, 0, stream>>>(w_inter, wconv);
  cvt_mlp_w_kernel<<<(CC * HID + NTHR - 1) / NTHR, NTHR, 0, stream>>>(w_mlp1, w_mlp2, w1c, w2c);
  // 1) LN1: x -> buf0 (f32)
  ln_kernel<false><<<BB * CC, NTHR, 0, stream>>>(x, (void*)buf0);
  // 2) attention: buf0 -> buf1 (bf16)
  attn_kernel<<<BB * CC, NTHR, attn_smem, stream>>>(buf0, wq, bq, wk, bk, wv, bv, buf1);
  // 3) 3x3 conv + residual: buf1 (+x) -> buf0 (= x_new)
  conv3x3_res_kernel<<<BB * HH, NTHR, conv_smem, stream>>>(buf1, x, wconv, buf0);
  // 4) LN2: buf0 -> xn2 (bf16, reusing buf1)
  ln_kernel<true><<<BB * CC, NTHR, 0, stream>>>(buf0, (void*)buf1);
  // 5) MLP + residual: -> d_out
  mlp_kernel<<<BB * HH, NTHR, mlp_smem, stream>>>(buf1, buf0, w1c, w2c, out);
}
